// Attention_45956150067314
// MI455X (gfx1250) — matrix operand order repaired and checked by reading, not compiled, not run on hardware
//
#include <hip/hip_runtime.h>
#include <hip/hip_bf16.h>

// ---------------------------------------------------------------------------
// Problem constants (from reference): B=8, C=192, Cq=32, H=W=128, N=16384
// ---------------------------------------------------------------------------
#define BB   8
#define CC   192
#define CQ   32
#define NN   16384
#define NT   128          // pixels per block tile
#define NTILES (NN / NT)  // 128 tiles per batch

typedef __bf16 bf16_t;
typedef __attribute__((ext_vector_type(16))) __bf16 v16bf;
typedef __attribute__((ext_vector_type(8)))  float  v8f;
typedef __attribute__((ext_vector_type(2)))  float  f32x2;
typedef __attribute__((ext_vector_type(2)))  __bf16 bf16x2;

struct B32x8 { uint4 a, b; };
struct U32x8 { unsigned u[8]; };
struct BF8   { bf16_t h[8]; };

__device__ __forceinline__ v16bf frag_from2(uint4 a, uint4 b) {
    B32x8 t; t.a = a; t.b = b;
    return __builtin_bit_cast(v16bf, t);
}

// A fragment (16x32 bf16) from a row-major bf16 matrix (global or LDS).
// lane half=lane>>4, M=lane&15; vgprs 0..7 = K kbase+half*16+0..15
// -> two contiguous 16B loads (same k order as B fragments).
__device__ __forceinline__ v16bf load_a_bf16(const bf16_t* __restrict__ row,
                                             int kbase, int half) {
    const bf16_t* p = row + kbase + half * 16;
    uint4 lo = *(const uint4*)(p);
    uint4 hi = *(const uint4*)(p + 8);
    return frag_from2(lo, hi);
}

// B fragment (32x16 bf16) from row-major-along-K bf16 (LDS).
// vgpr j = K pair half*16 + 2j -> 16 contiguous bf16 = two 16B loads.
__device__ __forceinline__ v16bf load_b_bf16(const bf16_t* __restrict__ row,
                                             int kbase, int half) {
    const bf16_t* p = row + kbase + half * 16;
    uint4 lo = *(const uint4*)(p);
    uint4 hi = *(const uint4*)(p + 8);
    return frag_from2(lo, hi);
}

// B fragment built from an f32 row in LDS with packed f32->bf16 conversion.
__device__ __forceinline__ v16bf load_b_cvt_f32(const float* __restrict__ row,
                                                int kbase, int half) {
    const float* q = row + kbase + half * 16;
    U32x8 t;
    #pragma unroll
    for (int j = 0; j < 8; ++j) {
        f32x2 v = *(const f32x2*)(q + 2 * j);
        bf16x2 c = __builtin_convertvector(v, bf16x2);
        t.u[j] = __builtin_bit_cast(unsigned, c);
    }
    return __builtin_bit_cast(v16bf, t);
}

// ---------------------------------------------------------------------------
// Kernel 0: zero f32 accumulators;  k_cvt: f32 -> bf16 (row-major copy)
// ---------------------------------------------------------------------------
__global__ void k0_zero(float* __restrict__ p, int n) {
    int i = blockIdx.x * blockDim.x + threadIdx.x;
    if (i < n) p[i] = 0.0f;
}
__global__ void k_cvt(const float* __restrict__ s, bf16_t* __restrict__ d, int n) {
    int i = blockIdx.x * blockDim.x + threadIdx.x;
    if (i < n) d[i] = (bf16_t)s[i];
}

// ---------------------------------------------------------------------------
// Kernel 1: Q/K projection from x1, L2-normalize over 32 channels,
//           write Qn/Kn (bf16 [B][32][N]), accumulate Ksum[b][32].
// LDS: x1 tile transposed, f32 [NT][XF_PAD]
// ---------------------------------------------------------------------------
#define XF_PAD 196                       // f32 row stride (784B, 16B-aligned)
#define K1_LDS (NT * XF_PAD * 4)
__global__ void k1_proj_qk(const float* __restrict__ x1,
                           const bf16_t* __restrict__ WqB, const float* __restrict__ bq,
                           const bf16_t* __restrict__ WkB, const float* __restrict__ bk,
                           bf16_t* __restrict__ Qn, bf16_t* __restrict__ Kn,
                           float* __restrict__ Ksum) {
    extern __shared__ char smem[];
    float* xT = (float*)smem;            // [NT][XF_PAD]

    const int b  = blockIdx.x / NTILES;
    const int n0 = (blockIdx.x % NTILES) * NT;
    const int tid = threadIdx.x;

    // stage: float4 along n (coalesced b128), scatter-transpose into LDS
    for (int idx = tid; idx < (CC * NT) / 4; idx += 256) {
        const int c  = idx >> 5;          // 0..191
        const int nq = (idx & 31) << 2;   // 0,4,..,124
        const float4 v = *(const float4*)(x1 + ((long)b * CC + c) * (long)NN + n0 + nq);
        float* dst = xT + nq * XF_PAD + c;
        dst[0] = v.x; dst[XF_PAD] = v.y; dst[2 * XF_PAD] = v.z; dst[3 * XF_PAD] = v.w;
    }
    __syncthreads();

    const int lane = tid & 31, w = tid >> 5;
    const int half = lane >> 4, lq = lane & 15;
    const float* xrow = xT + (w * 16 + lq) * XF_PAD;

    v8f aq0 = {}, aq1 = {}, ak0 = {}, ak1 = {};
    #pragma unroll
    for (int kc = 0; kc < 6; ++kc) {
        v16bf bx  = load_b_cvt_f32(xrow, kc * 32, half);
        v16bf aQ0 = load_a_bf16(WqB + (0  + lq) * CC, kc * 32, half);
        v16bf aQ1 = load_a_bf16(WqB + (16 + lq) * CC, kc * 32, half);
        v16bf aK0 = load_a_bf16(WkB + (0  + lq) * CC, kc * 32, half);
        v16bf aK1 = load_a_bf16(WkB + (16 + lq) * CC, kc * 32, half);
        aq0 = __builtin_amdgcn_wmma_f32_16x16x32_bf16(false, aQ0, false, bx, (short)0, aq0, false, false);
        aq1 = __builtin_amdgcn_wmma_f32_16x16x32_bf16(false, aQ1, false, bx, (short)0, aq1, false, false);
        ak0 = __builtin_amdgcn_wmma_f32_16x16x32_bf16(false, aK0, false, bx, (short)0, ak0, false, false);
        ak1 = __builtin_amdgcn_wmma_f32_16x16x32_bf16(false, aK1, false, bx, (short)0, ak1, false, false);
    }

    // bias + per-column L2 norm over the 32 channels
    float sq = 0.0f, sk = 0.0f;
    #pragma unroll
    for (int g = 0; g < 8; ++g) {
        const int mA = g + 8 * half, mB = 16 + mA;
        aq0[g] += bq[mA]; aq1[g] += bq[mB];
        ak0[g] += bk[mA]; ak1[g] += bk[mB];
        sq += aq0[g] * aq0[g] + aq1[g] * aq1[g];
        sk += ak0[g] * ak0[g] + ak1[g] * ak1[g];
    }
    sq += __shfl_xor(sq, 16, 32);
    sk += __shfl_xor(sk, 16, 32);
    const float rq = rsqrtf(sq), rk = rsqrtf(sk);

    const long base = (long)b * CQ * NN;
    const int  ncol = n0 + w * 16 + lq;
    #pragma unroll
    for (int g = 0; g < 8; ++g) {
        const int mA = g + 8 * half, mB = 16 + mA;
        Qn[base + (long)mA * NN + ncol] = (bf16_t)(aq0[g] * rq);
        Qn[base + (long)mB * NN + ncol] = (bf16_t)(aq1[g] * rq);
        const float kA = ak0[g] * rk, kB = ak1[g] * rk;
        Kn[base + (long)mA * NN + ncol] = (bf16_t)kA;
        Kn[base + (long)mB * NN + ncol] = (bf16_t)kB;
        float vA = kA, vB = kB;
        #pragma unroll
        for (int msk = 1; msk < 16; msk <<= 1) {
            vA += __shfl_xor(vA, msk, 32);
            vB += __shfl_xor(vB, msk, 32);
        }
        if (lq == 0) {
            atomicAdd(&Ksum[b * CQ + mA], vA);
            atomicAdd(&Ksum[b * CQ + mB], vB);
        }
    }
}

// ---------------------------------------------------------------------------
// Kernel 2: V on the fly from x; matrix[b] += Kn_tile · V_tileᵀ; xsum += Σn x
// LDS: xT f32 [NT][XF_PAD] | Vl bf16 [CC][VL_PAD] | KnL bf16 [CQ][KN_PAD]
// ---------------------------------------------------------------------------
#define VL_PAD 136
#define KN_PAD 136
#define K2_LDS (NT * XF_PAD * 4 + CC * VL_PAD * 2 + CQ * KN_PAD * 2)
__global__ void k2_v_matrix(const float* __restrict__ x,
                            const bf16_t* __restrict__ WvB, const float* __restrict__ bv,
                            const bf16_t* __restrict__ Kn,
                            float* __restrict__ matrixWS, float* __restrict__ xsumWS) {
    extern __shared__ char smem[];
    float*  xT  = (float*)smem;                                        // [NT][XF_PAD]
    bf16_t* Vl  = (bf16_t*)(smem + NT * XF_PAD * 4);                   // [CC][VL_PAD]
    bf16_t* KnL = (bf16_t*)(smem + NT * XF_PAD * 4 + CC * VL_PAD * 2); // [CQ][KN_PAD]

    const int b  = blockIdx.x / NTILES;
    const int n0 = (blockIdx.x % NTILES) * NT;
    const int tid = threadIdx.x;

    for (int idx = tid; idx < (CC * NT) / 4; idx += 256) {
        const int c  = idx >> 5;
        const int nq = (idx & 31) << 2;
        const float4 v = *(const float4*)(x + ((long)b * CC + c) * (long)NN + n0 + nq);
        float* dst = xT + nq * XF_PAD + c;
        dst[0] = v.x; dst[XF_PAD] = v.y; dst[2 * XF_PAD] = v.z; dst[3 * XF_PAD] = v.w;
    }
    for (int idx = tid; idx < (CQ * NT) / 8; idx += 256) {   // untransposed copy
        const int m  = idx >> 4;
        const int nq = (idx & 15) << 3;
        *(uint4*)(KnL + m * KN_PAD + nq) =
            *(const uint4*)(Kn + ((long)b * CQ + m) * (long)NN + n0 + nq);
    }
    __syncthreads();

    if (tid < CC) {                      // xsum partial for value_sum GEMV
        float s = 0.0f;
        for (int nn = 0; nn < NT; ++nn) s += xT[nn * XF_PAD + tid];
        atomicAdd(&xsumWS[b * CC + tid], s);
    }

    const int lane = tid & 31, w = tid >> 5;
    const int half = lane >> 4, lq = lane & 15;
    const float* xrow = xT + (w * 16 + lq) * XF_PAD;

    // ---- V tile: wave w owns 16 columns, all 12 c-tiles -------------------
    v8f acc[12];
    #pragma unroll
    for (int ct = 0; ct < 12; ++ct) { v8f z = {}; acc[ct] = z; }
    #pragma unroll
    for (int kc = 0; kc < 6; ++kc) {
        v16bf bx = load_b_cvt_f32(xrow, kc * 32, half);
        #pragma unroll
        for (int ct = 0; ct < 12; ++ct) {
            v16bf aw = load_a_bf16(WvB + (ct * 16 + lq) * CC, kc * 32, half);
            acc[ct] = __builtin_amdgcn_wmma_f32_16x16x32_bf16(false, aw, false, bx, (short)0, acc[ct], false, false);
        }
    }
    #pragma unroll
    for (int ct = 0; ct < 12; ++ct) {    // bias + stash V bf16 as [c][n]
        #pragma unroll
        for (int g = 0; g < 8; ++g) {
            const int c = ct * 16 + g + 8 * half;
            Vl[c * VL_PAD + w * 16 + lq] = (bf16_t)(acc[ct][g] + bv[c]);
        }
    }
    __syncthreads();

    // ---- matrix[m][c] += Σn Kn[m][n]·V[c][n] : 24 tiles over 8 waves ------
    #pragma unroll
    for (int tt = 0; tt < 3; ++tt) {
        const int t  = w + tt * 8;
        const int mt = t / 12, ct = t % 12;
        v8f a2 = {};
        #pragma unroll
        for (int nc = 0; nc < 4; ++nc) {
            v16bf aK = load_a_bf16(KnL + (mt * 16 + lq) * KN_PAD, nc * 32, half);
            v16bf bV = load_b_bf16(Vl + (ct * 16 + lq) * VL_PAD, nc * 32, half);
            a2 = __builtin_amdgcn_wmma_f32_16x16x32_bf16(false, aK, false, bV, (short)0, a2, false, false);
        }
        #pragma unroll
        for (int g = 0; g < 8; ++g) {
            const int m = mt * 16 + g + 8 * half;
            const int c = ct * 16 + lq;
            atomicAdd(&matrixWS[((long)b * CQ + m) * CC + c], a2[g]);
        }
    }
}

// ---------------------------------------------------------------------------
// Kernel 2b: value_sum[b][c] = Wv[c,:]·xsum[b,:] + N*bv[c]
// Kernel 2c: matTBF[b][c][m] = (bf16) matrixWS[b][m][c]
// ---------------------------------------------------------------------------
__global__ void k2b_vsum(const float* __restrict__ Wv, const float* __restrict__ bv,
                         const float* __restrict__ xsumWS, float* __restrict__ vsumWS) {
    const int b = blockIdx.x, c = threadIdx.x;
    if (c >= CC) return;
    float s = 0.0f;
    for (int k = 0; k < CC; ++k) s += Wv[c * CC + k] * xsumWS[b * CC + k];
    vsumWS[b * CC + c] = s + (float)NN * bv[c];
}
__global__ void k2c_matT(const float* __restrict__ matrixWS, bf16_t* __restrict__ matT) {
    int i = blockIdx.x * blockDim.x + threadIdx.x;
    if (i >= BB * CQ * CC) return;
    const int b = i / (CQ * CC), r = i % (CQ * CC);
    const int m = r / CC, c = r % CC;
    matT[(long)b * CC * CQ + c * CQ + m] = (bf16_t)matrixWS[i];
}

// ---------------------------------------------------------------------------
// Kernel 3: out[c,n] = gamma * tailor[n] * (value_sum[c] + Σm Qn[m,n]·matrix[m,c])
// ---------------------------------------------------------------------------
#define QT_PAD 40
#define K3_LDS (NT * QT_PAD * 2 + NT * 4 + CC * 4 + CQ * 4)
__global__ void k3_out(const bf16_t* __restrict__ Qn,
                       const bf16_t* __restrict__ matT,   // [B][192][32] bf16
                       const float* __restrict__ vsumWS,
                       const float* __restrict__ KsumWS,
                       const float* __restrict__ gamma,
                       float* __restrict__ out) {
    extern __shared__ char smem[];
    bf16_t* QnT     = (bf16_t*)smem;                    // [NT][QT_PAD]
    float*  tailorL = (float*)(smem + NT * QT_PAD * 2); // [NT]
    float*  vsumL   = tailorL + NT;                     // [CC]
    float*  KsumL   = vsumL + CC;                       // [CQ]

    const int b  = blockIdx.x / NTILES;
    const int n0 = (blockIdx.x % NTILES) * NT;
    const int tid = threadIdx.x;

    for (int idx = tid; idx < (CQ * NT) / 8; idx += 256) {
        const int m  = idx >> 4;
        const int nq = (idx & 15) << 3;
        uint4 v = *(const uint4*)(Qn + ((long)b * CQ + m) * (long)NN + n0 + nq);
        BF8 t = __builtin_bit_cast(BF8, v);
        #pragma unroll
        for (int i = 0; i < 8; ++i) QnT[(nq + i) * QT_PAD + m] = t.h[i];
    }
    if (tid < CC) vsumL[tid] = vsumWS[b * CC + tid];
    if (tid < CQ) KsumL[tid] = KsumWS[b * CQ + tid] + 1e-6f;   // Ksum + EPS
    __syncthreads();

    if (tid < NT) {
        float s = 0.0f;
        #pragma unroll
        for (int m = 0; m < CQ; ++m) s += (float)QnT[tid * QT_PAD + m] * KsumL[m];
        tailorL[tid] = 1.0f / ((float)NN + s);
    }
    __syncthreads();

    const int lane = tid & 31, w = tid >> 5;
    const int half = lane >> 4, lq = lane & 15;
    const float gam = gamma[0];

    // B fragment (K=m, N=n) shared across the 12 c-tiles of this wave
    const int nloc = w * 16 + lq;
    v16bf bQ = load_b_bf16(QnT + nloc * QT_PAD, 0, half);

    #pragma unroll
    for (int ct = 0; ct < 12; ++ct) {
        v16bf aM = load_a_bf16(matT + ((long)b * CC + ct * 16 + lq) * CQ, 0, half);
        v8f a3 = {};
        a3 = __builtin_amdgcn_wmma_f32_16x16x32_bf16(false, aM, false, bQ, (short)0, a3, false, false);
        #pragma unroll
        for (int g = 0; g < 8; ++g) {
            const int cc = ct * 16 + g + 8 * half;
            out[((long)b * CC + cc) * (long)NN + n0 + nloc] =
                gam * tailorL[nloc] * (vsumL[cc] + a3[g]);
        }
    }
}

// ---------------------------------------------------------------------------
// Launch
// ---------------------------------------------------------------------------
extern "C" void kernel_launch(void* const* d_in, const int* in_sizes, int n_in,
                              void* d_out, int out_size, void* d_ws, size_t ws_size,
                              hipStream_t stream) {
    const float* x     = (const float*)d_in[0];
    const float* x1    = (const float*)d_in[1];
    const float* Wq    = (const float*)d_in[2];
    const float* bq    = (const float*)d_in[3];
    const float* Wk    = (const float*)d_in[4];
    const float* bk    = (const float*)d_in[5];
    const float* Wv    = (const float*)d_in[6];
    const float* bv    = (const float*)d_in[7];
    const float* gamma = (const float*)d_in[8];
    float* outp = (float*)d_out;

    // ---- workspace layout -------------------------------------------------
    float* wsf      = (float*)d_ws;
    float* matrixWS = wsf;                    // B*32*192 = 49152 f32
    float* KsumWS   = wsf + 49152;            // 256
    float* xsumWS   = wsf + 49408;            // 1536
    float* vsumWS   = wsf + 50944;            // 1536
    bf16_t* bfbase = (bf16_t*)((char*)d_ws + 212992);       // 16B-aligned
    bf16_t* QnWS = bfbase;                                  // 8*32*16384
    bf16_t* KnWS = QnWS + (size_t)BB * CQ * NN;
    bf16_t* WqBF = KnWS + (size_t)BB * CQ * NN;             // 32*192
    bf16_t* WkBF = WqBF + CQ * CC;
    bf16_t* WvBF = WkBF + CQ * CC;                          // 192*192
    bf16_t* matT = WvBF + CC * CC;                          // B*192*32

    (void)hipFuncSetAttribute((const void*)k1_proj_qk,
                              hipFuncAttributeMaxDynamicSharedMemorySize, K1_LDS);
    (void)hipFuncSetAttribute((const void*)k2_v_matrix,
                              hipFuncAttributeMaxDynamicSharedMemorySize, K2_LDS);

    const int nzero = 49152 + 256 + 1536;     // matrix + Ksum + xsum
    k0_zero<<<(nzero + 255) / 256, 256, 0, stream>>>(wsf, nzero);
    k_cvt<<<(CQ * CC + 255) / 256, 256, 0, stream>>>(Wq, WqBF, CQ * CC);
    k_cvt<<<(CQ * CC + 255) / 256, 256, 0, stream>>>(Wk, WkBF, CQ * CC);
    k_cvt<<<(CC * CC + 255) / 256, 256, 0, stream>>>(Wv, WvBF, CC * CC);

    dim3 grid(BB * NTILES), blk(256);
    k1_proj_qk<<<grid, blk, K1_LDS, stream>>>(x1, WqBF, bq, WkBF, bk, QnWS, KnWS, KsumWS);
    k2_v_matrix<<<grid, blk, K2_LDS, stream>>>(x, WvBF, bv, KnWS, matrixWS, xsumWS);
    k2b_vsum<<<BB, 192, 0, stream>>>(Wv, bv, xsumWS, vsumWS);
    k2c_matT<<<(BB * CQ * CC + 255) / 256, 256, 0, stream>>>(matrixWS, matT);
    k3_out<<<grid, blk, K3_LDS, stream>>>(QnWS, matT, vsumWS, KsumWS, gamma, outp);
}
